// Arrow_Attention_44169443672539
// MI455X (gfx1250) — matrix operand order repaired and checked by reading, not compiled, not run on hardware
//
#include <hip/hip_runtime.h>

// ---------------- problem constants ----------------
#define BB   4
#define SS   4096
#define HIDD 768
#define NHH  12
#define DDIM 64
#define WWIN 128
#define CCH  (SS / WWIN)     // 32 chunks
#define BHH  (BB * NHH)      // 48
#define MTOT (BB * SS)       // 16384

typedef __attribute__((ext_vector_type(16))) __bf16 v16bf;
typedef __attribute__((ext_vector_type(8)))  __bf16 v8bf;
typedef __attribute__((ext_vector_type(4)))  __bf16 v4bf;
typedef __attribute__((ext_vector_type(8)))  float  v8f;
typedef __attribute__((ext_vector_type(4)))  unsigned int u32x4;
typedef __attribute__((ext_vector_type(4)))  int i32x4;
typedef __attribute__((ext_vector_type(8)))  int i32x8;

#if defined(__has_builtin)
#if __has_builtin(__builtin_amdgcn_tensor_load_to_lds) && \
    __has_builtin(__builtin_amdgcn_s_wait_tensorcnt)
#define USE_TDM 1
#endif
#endif
#ifndef USE_TDM
#define USE_TDM 0
#endif

__device__ inline v8f zero8() {
  v8f r;
#pragma unroll
  for (int i = 0; i < 8; ++i) r[i] = 0.0f;
  return r;
}

__device__ inline v16bf combine16(v8bf lo, v8bf hi) {
  v16bf r;
#pragma unroll
  for (int i = 0; i < 8; ++i) { r[i] = lo[i]; r[i + 8] = hi[i]; }
  return r;
}

__device__ inline v8f wmma_bf16(v16bf a, v16bf b, v8f c) {
  return __builtin_amdgcn_wmma_f32_16x16x32_bf16(false, a, false, b,
                                                 (short)0, c, false, false);
}

#if USE_TDM
// TDM: DMA a 2D bf16 tile (tile_d0 contiguous elems per row, tile_d1 rows,
// row stride `stride0` elems) from global memory into LDS at byte offset
// lds_off.  Descriptor packing per CDNA5 ISA sec. 8.3/8.4 (D# groups 0/1).
__device__ inline void tdm_load_2d_bf16(unsigned lds_off, const void* gptr,
                                        unsigned tile_d0, unsigned tile_d1,
                                        unsigned long long stride0) {
  unsigned long long ga = (unsigned long long)gptr;
  u32x4 g0;
  g0[0] = 1u;                                   // count=1, user mode, no gather
  g0[1] = lds_off;                              // lds_addr (bytes)
  g0[2] = (unsigned)(ga & 0xffffffffu);         // global_addr[31:0]
  g0[3] = (unsigned)((ga >> 32) & 0x01ffffffu)  // global_addr[56:32]
          | (2u << 30);                         // type = 2 ("image")
  unsigned td0 = tile_d0, td1 = tile_d1;        // tensor dims == tile dims
  i32x8 g1;
  g1[0] = (int)(1u << 16);                      // wg_mask=0, data_size=1 (2B)
  g1[1] = (int)((td0 & 0xffffu) << 16);         // tensor_dim0[15:0]
  g1[2] = (int)(((td0 >> 16) & 0xffffu) | ((td1 & 0xffffu) << 16));
  g1[3] = (int)(((td1 >> 16) & 0xffffu) | ((tile_d0 & 0xffffu) << 16));
  g1[4] = (int)(tile_d1 & 0xffffu);             // tile_dim1; tile_dim2=0
  g1[5] = (int)(unsigned)(stride0 & 0xffffffffu);      // dim0_stride[31:0]
  g1[6] = (int)(unsigned)((stride0 >> 32) & 0xffffu);  // dim0_stride[47:32]
  g1[7] = 0;
  i32x4 gz = {0, 0, 0, 0};
#if __clang_major__ >= 23
  i32x8 gz8 = {0, 0, 0, 0, 0, 0, 0, 0};
  __builtin_amdgcn_tensor_load_to_lds(g0, g1, gz, gz, gz8, 0);
#else
  __builtin_amdgcn_tensor_load_to_lds(g0, g1, gz, gz, 0);
#endif
}
__device__ inline unsigned lds_offset_of(const void* p) {
  return (unsigned)(unsigned long long)p;       // low 32 bits = LDS offset
}
#endif

// ---------------- fp32 -> bf16 conversions ----------------
__global__ __launch_bounds__(256) void cvt_x(const float* __restrict__ x,
                                             __bf16* __restrict__ y, int n4) {
  int i = blockIdx.x * blockDim.x + threadIdx.x;
  if (i >= n4) return;
  float4 v = ((const float4*)x)[i];
  v4bf o;
  o[0] = (__bf16)v.x; o[1] = (__bf16)v.y; o[2] = (__bf16)v.z; o[3] = (__bf16)v.w;
  ((v4bf*)y)[i] = o;
}

__global__ __launch_bounds__(256) void cvt_w(const float* __restrict__ wq,
                                             const float* __restrict__ wk,
                                             const float* __restrict__ wv,
                                             __bf16* __restrict__ out) {
  int i = blockIdx.x * blockDim.x + threadIdx.x;
  const int n = HIDD * HIDD;
  if (i >= n) return;
  out[i]         = (__bf16)wq[i];
  out[n + i]     = (__bf16)wk[i];
  out[2 * n + i] = (__bf16)wv[i];
}

// ---------------- QKV projection GEMM (bf16 WMMA, f32 accum) ----------------
// grid: (MTOT/32, HIDD/256, 3)   block: 256 (8 waves, 2M x 4N tiling)
__global__ __launch_bounds__(256) void qkv_gemm(
    const __bf16* __restrict__ X, const __bf16* __restrict__ Wall,
    const float* __restrict__ bq, const float* __restrict__ bk,
    const float* __restrict__ bv,
    __bf16* __restrict__ qb, __bf16* __restrict__ kb, __bf16* __restrict__ vT) {
  const int lane = threadIdx.x & 31;
  const int wave = threadIdx.x >> 5;
  const int hi   = lane >> 4;
  const int l15  = lane & 15;
  const int z    = blockIdx.z;

  const int mtile = blockIdx.x * 32 + (wave >> 2) * 16;
  const int nbase = blockIdx.y * 256 + (wave & 3) * 64;

  const __bf16* Wz   = Wall + (size_t)z * HIDD * HIDD;
  const float*  bias = (z == 0) ? bq : ((z == 1) ? bk : bv);

  v8f acc[4];
#pragma unroll
  for (int t = 0; t < 4; ++t) acc[t] = zero8();

  const __bf16* arow = X + (size_t)(mtile + l15) * HIDD;
  for (int k0 = 0; k0 < HIDD; k0 += 32) {
    v8bf alo = *(const v8bf*)(arow + k0 + hi * 16);
    v8bf ahi = *(const v8bf*)(arow + k0 + hi * 16 + 8);
    v16bf a  = combine16(alo, ahi);
#pragma unroll
    for (int t = 0; t < 4; ++t) {
      int n   = nbase + t * 16 + l15;
      v16bf b = *(const v16bf*)(Wz + (size_t)n * HIDD + k0 + hi * 16);
      acc[t]  = wmma_bf16(a, b, acc[t]);
    }
  }

#pragma unroll
  for (int t = 0; t < 4; ++t) {
    int nn = nbase + t * 16 + l15;
    int hh = nn >> 6, dd = nn & 63;
    float bval = bias[nn];
    if (z == 2) {
      v8bf pv;
#pragma unroll
      for (int r = 0; r < 8; ++r) pv[r] = (__bf16)(acc[t][r] + bval);
      int m0 = mtile + hi * 8;
      int b0 = m0 >> 12, s0 = m0 & (SS - 1);
      *(v8bf*)(vT + ((size_t)(b0 * NHH + hh) * DDIM + dd) * SS + s0) = pv;
    } else {
      __bf16* dst = (z == 0) ? qb : kb;
      float scale = (z == 0) ? 0.125f : 1.0f;  // 1/sqrt(64) folded into q
#pragma unroll
      for (int r = 0; r < 8; ++r) {
        int m  = mtile + hi * 8 + r;
        int b0 = m >> 12, s0 = m & (SS - 1);
        float val = (acc[t][r] + bval) * scale;
        dst[((size_t)(b0 * NHH + hh) * SS + s0) * DDIM + dd] = (__bf16)val;
      }
    }
  }
}

// ---------------- sliding-window attention (flash-style + TDM) ----------------
// grid: (BHH, CCH) block: 256.  Each wave owns a 16-query M-tile of the chunk.
// K/V tiles of each 128-key block are DMA'd into LDS (double-buffered) by the
// Tensor Data Mover while the previous block's WMMAs run.
__global__ __launch_bounds__(256) void swattn(
    const __bf16* __restrict__ q, const __bf16* __restrict__ k,
    const __bf16* __restrict__ vT, float* __restrict__ out) {
  __shared__ __attribute__((aligned(64))) __bf16 Kls[2][WWIN * DDIM];  // 2x16KB
  __shared__ __attribute__((aligned(64))) __bf16 Vls[2][DDIM * WWIN];  // 2x16KB
  __shared__ __attribute__((aligned(64))) __bf16 plds[8][16][WWIN];    // 32KB

  const int lane = threadIdx.x & 31;
  const int wave = threadIdx.x >> 5;
  const int hi   = lane >> 4;
  const int l15  = lane & 15;
  const int bh   = blockIdx.x;
  const int c    = blockIdx.y;
  const int bidx = bh / NHH, head = bh % NHH;
  const int mbase = c * WWIN + wave * 16;

  const __bf16* kbh = k + (size_t)bh * SS * DDIM;
  const __bf16* vbh = vT + (size_t)bh * DDIM * SS;

  // q A-fragments for both 32-wide K-steps (D = 64), kept in registers
  const __bf16* qrow = q + ((size_t)bh * SS + mbase + l15) * DDIM;
  v16bf qa[2];
#pragma unroll
  for (int ks = 0; ks < 2; ++ks) {
    v8bf lo = *(const v8bf*)(qrow + ks * 32 + hi * 16);
    v8bf h8 = *(const v8bf*)(qrow + ks * 32 + hi * 16 + 8);
    qa[ks]  = combine16(lo, h8);
  }

  v8f ctx[4];
#pragma unroll
  for (int t = 0; t < 4; ++t) ctx[t] = zero8();
  float mrow[8], lrow[8];
#pragma unroll
  for (int r = 0; r < 8; ++r) { mrow[r] = -INFINITY; lrow[r] = 0.0f; }

  // stage key-block j (kb0 = (c-1+j)*W) into LDS buffer `buf`
  auto stage = [&](int j, int buf) {
    const int kb0 = (c - 1 + j) * WWIN;
#if USE_TDM
    tdm_load_2d_bf16(lds_offset_of(&Kls[buf][0]), kbh + (size_t)kb0 * DDIM,
                     DDIM, WWIN, DDIM);                 // 128 rows x 64, stride 64
    tdm_load_2d_bf16(lds_offset_of(&Vls[buf][0]), vbh + kb0,
                     WWIN, DDIM, SS);                   // 64 rows x 128, stride S
#else
    for (int t = lane; t < (WWIN * DDIM) / 8; t += 32)
      ((v8bf*)&Kls[buf][0])[t] = *(const v8bf*)(kbh + (size_t)kb0 * DDIM + t * 8);
    for (int t = lane; t < DDIM * (WWIN / 8); t += 32) {
      int d = t >> 4, cb = (t & 15) * 8;
      ((v8bf*)&Vls[buf][0])[d * (WWIN / 8) + (t & 15)] =
          *(const v8bf*)(vbh + (size_t)d * SS + kb0 + cb);
    }
#endif
  };

  const int jfirst = (c == 0) ? 1 : 0;          // boundary blocks fully masked
  const int jlast  = (c == CCH - 1) ? 1 : 2;
  const int nb     = jlast - jfirst + 1;

  if (wave == 0) stage(jfirst, 0);

  for (int idx = 0; idx < nb; ++idx) {
    const int j   = jfirst + idx;
    const int buf = idx & 1;
    if (wave == 0) {
      if (idx + 1 < nb) {
        stage(j + 1, buf ^ 1);
#if USE_TDM
        __builtin_amdgcn_s_wait_tensorcnt(2);   // current block's 2 DMAs done
#endif
      } else {
#if USE_TDM
        __builtin_amdgcn_s_wait_tensorcnt(0);
#endif
      }
    }
    __syncthreads();                             // LDS tile `buf` published

    // ---- scores: q (16xD) @ K_block^T (Dx128) -> 8 N-tiles of 16x16 ----
    const __bf16* Kb = &Kls[buf][0];
    v8f sc[8];
#pragma unroll
    for (int nt = 0; nt < 8; ++nt) sc[nt] = zero8();
#pragma unroll
    for (int ks = 0; ks < 2; ++ks) {
#pragma unroll
      for (int nt = 0; nt < 8; ++nt) {
        v16bf bf = *(const v16bf*)(Kb + (nt * 16 + l15) * DDIM + ks * 32 + hi * 16);
        sc[nt]   = wmma_bf16(qa[ks], bf, sc[nt]);
      }
    }

    // ---- band mask: prev block -> upper triangle, next block -> lower ----
    if (j != 1) {
#pragma unroll
      for (int nt = 0; nt < 8; ++nt) {
        int np = nt * 16 + l15;
#pragma unroll
        for (int r = 0; r < 8; ++r) {
          int mp = wave * 16 + hi * 8 + r;
          bool valid = (j == 0) ? (np >= mp) : (np <= mp);
          if (!valid) sc[nt][r] = -INFINITY;
        }
      }
    }

    // ---- online softmax (rows live in 16-lane halves) ----
    float csc[8];
#pragma unroll
    for (int r = 0; r < 8; ++r) {
      float t = sc[0][r];
#pragma unroll
      for (int nt = 1; nt < 8; ++nt) t = fmaxf(t, sc[nt][r]);
      t = fmaxf(t, __shfl_xor(t, 1, 32));
      t = fmaxf(t, __shfl_xor(t, 2, 32));
      t = fmaxf(t, __shfl_xor(t, 4, 32));
      t = fmaxf(t, __shfl_xor(t, 8, 32));
      float mnew  = fmaxf(mrow[r], t);
      float scale = __expf(mrow[r] - mnew);
      float ssum  = 0.0f;
#pragma unroll
      for (int nt = 0; nt < 8; ++nt) {
        float p   = __expf(sc[nt][r] - mnew);
        sc[nt][r] = p;
        ssum     += p;
      }
      ssum += __shfl_xor(ssum, 1, 32);
      ssum += __shfl_xor(ssum, 2, 32);
      ssum += __shfl_xor(ssum, 4, 32);
      ssum += __shfl_xor(ssum, 8, 32);
      lrow[r] = lrow[r] * scale + ssum;
      mrow[r] = mnew;
      csc[r]  = scale;
    }
#pragma unroll
    for (int t = 0; t < 4; ++t)
#pragma unroll
      for (int r = 0; r < 8; ++r) ctx[t][r] *= csc[r];

    // ---- re-stripe P: C-layout -> LDS row-major -> A-layout ----
    __bf16* pl = &plds[wave][0][0];
#pragma unroll
    for (int nt = 0; nt < 8; ++nt)
#pragma unroll
      for (int r = 0; r < 8; ++r)
        pl[(hi * 8 + r) * WWIN + nt * 16 + l15] = (__bf16)sc[nt][r];
    __syncthreads();

    // ---- ctx += P (16x128) @ V_block (128x64) ----
    const __bf16* prow = &plds[wave][l15][0];
    const __bf16* Vb   = &Vls[buf][0];
#pragma unroll
    for (int kk = 0; kk < 4; ++kk) {
      v8bf lo = *(const v8bf*)(prow + kk * 32 + hi * 16);
      v8bf h8 = *(const v8bf*)(prow + kk * 32 + hi * 16 + 8);
      v16bf pa = combine16(lo, h8);
#pragma unroll
      for (int dt = 0; dt < 4; ++dt) {
        v16bf vb = *(const v16bf*)(Vb + (dt * 16 + l15) * WWIN + kk * 32 + hi * 16);
        ctx[dt]  = wmma_bf16(pa, vb, ctx[dt]);
      }
    }
    __syncthreads();                             // LDS buffers reusable
  }

  // ---- epilogue: normalize, scatter to [B,S,HID] fp32 ----
#pragma unroll
  for (int dt = 0; dt < 4; ++dt) {
#pragma unroll
    for (int r = 0; r < 8; ++r) {
      int s    = mbase + hi * 8 + r;
      float vv = ctx[dt][r] / lrow[r];
      out[((size_t)bidx * SS + s) * HIDD + head * DDIM + dt * 16 + l15] = vv;
    }
  }
}

// ---------------- host launcher ----------------
extern "C" void kernel_launch(void* const* d_in, const int* in_sizes, int n_in,
                              void* d_out, int out_size, void* d_ws, size_t ws_size,
                              hipStream_t stream) {
  (void)in_sizes; (void)n_in; (void)out_size; (void)ws_size;
  const float* hs = (const float*)d_in[0];
  const float* Wq = (const float*)d_in[1];
  const float* bq = (const float*)d_in[2];
  const float* Wk = (const float*)d_in[3];
  const float* bk = (const float*)d_in[4];
  const float* Wv = (const float*)d_in[5];
  const float* bv = (const float*)d_in[6];

  __bf16* Xbf = (__bf16*)d_ws;                          // [MTOT, HID]
  __bf16* Wbf = Xbf + (size_t)MTOT * HIDD;              // [3, HID, HID]
  __bf16* qb  = Wbf + (size_t)3 * HIDD * HIDD;          // [BH, S, D] (pre-scaled)
  __bf16* kb  = qb + (size_t)BHH * SS * DDIM;           // [BH, S, D]
  __bf16* vT  = kb + (size_t)BHH * SS * DDIM;           // [BH, D, S]

  {
    int n4 = (MTOT * HIDD) / 4;
    cvt_x<<<(n4 + 255) / 256, 256, 0, stream>>>(hs, Xbf, n4);
  }
  {
    int n = HIDD * HIDD;
    cvt_w<<<(n + 255) / 256, 256, 0, stream>>>(Wq, Wk, Wv, Wbf);
  }
  qkv_gemm<<<dim3(MTOT / 32, HIDD / 256, 3), 256, 0, stream>>>(
      Xbf, Wbf, bq, bk, bv, qb, kb, vT);
  swattn<<<dim3(BHH, CCH), 256, 0, stream>>>(qb, kb, vT, (float*)d_out);
}
